// GCNNode_37056977830250
// MI455X (gfx1250) — matrix operand order repaired and checked by reading, not compiled, not run on hardware
//
#include <hip/hip_runtime.h>

// GCN (3x GraphConv + BN + ReLU) + linear classifier for MI455X (gfx1250).
// Heavy part = edge scatter (memory/atomic bound, L2-resident feature mats).
// Dense GEMMs: v_wmma_f32_16x16x32_bf16, one wave per 16x64 output strip,
// B operand pre-packed to a contiguous-k lane layout (k = 16*(lane>>4) + e) so each lane does one 32B load.

typedef __attribute__((ext_vector_type(16))) __bf16 v16bf;
typedef __attribute__((ext_vector_type(8)))  float  v8f;

#define KSTEPS 4   // K = 128 = 4 x 32

// ---------------- utility kernels ----------------

__global__ void zero_f4(float* __restrict__ p, long long n4) {
  long long i = (long long)blockIdx.x * blockDim.x + threadIdx.x;
  if (i < n4) ((float4*)p)[i] = make_float4(0.f, 0.f, 0.f, 0.f);
}

__global__ void degree_k(const int* __restrict__ src, const int* __restrict__ dst,
                         float* __restrict__ degS, float* __restrict__ degD, int E) {
  int e = blockIdx.x * blockDim.x + threadIdx.x;
  if (e >= E) return;
  atomicAdd(&degS[src[e]], 1.0f);
  atomicAdd(&degD[dst[e]], 1.0f);
}

__global__ void dinv_k(float* __restrict__ d, int n) {
  int i = blockIdx.x * blockDim.x + threadIdx.x;
  if (i >= n) return;
  float v = d[i];
  d[i] = (v > 0.f) ? rsqrtf(fmaxf(v, 1.0f)) : 0.f;
}

// ---------------- SpMM scatter: one wave (32 lanes) per edge ----------------
// m[dst] += dinv_src[src]*dinv_dst[dst] * h[src]  (128 feats = 32 lanes x float4)

__global__ void scatter_k(const float* __restrict__ h, const int* __restrict__ src,
                          const int* __restrict__ dst, const float* __restrict__ dinvS,
                          const float* __restrict__ dinvD, float* __restrict__ M, int E) {
  int e = (int)(((long long)blockIdx.x * blockDim.x + threadIdx.x) >> 5);
  if (e >= E) return;
  int lane = threadIdx.x & 31;
  int s = src[e], d = dst[e];
  float c = dinvS[s] * dinvD[d];
  float4 v = ((const float4*)(h + (size_t)s * 128))[lane];
  float* mp = M + (size_t)d * 128 + lane * 4;
  atomicAdd(mp + 0, c * v.x);
  atomicAdd(mp + 1, c * v.y);
  atomicAdd(mp + 2, c * v.z);
  atomicAdd(mp + 3, c * v.w);
}

// ---------------- pack W[128 x ncols] (f32, row-major) into WMMA B layout (bf16) ----
// Bp[((tn*KSTEPS + ks)*32 + lane)*16 + e] = bf16( W[(ks*32 + (lane>>4)*16 + e)*ncols
//                                                  + tn*16 + (lane&15)] )
// -> per (tile, K-step) a wave's B fragment is one contiguous 32B chunk per lane.

__global__ void packB_k(const float* __restrict__ W, __bf16* __restrict__ Bp, int ncols) {
  int idx = blockIdx.x * blockDim.x + threadIdx.x;
  int total = (ncols >> 4) * KSTEPS * 32 * 16;
  if (idx >= total) return;
  int e    = idx & 15;
  int lane = (idx >> 4) & 31;
  int ks   = (idx >> 9) & 3;
  int tn   = idx >> 11;
  int K    = ks * 32 + (lane >> 4) * 16 + e;
  int col  = tn * 16 + (lane & 15);
  Bp[idx] = (__bf16)W[(size_t)K * ncols + col];
}

// ---------------- WMMA GEMM: Y[nrows x ncols] = X[nrows x 128] @ W[128 x ncols] ----
// One wave per 16 x (16*CT) output strip: A fragment loaded once per K-step,
// reused for CT column tiles (16 WMMAs per wave for CT=4).
// A layout (contiguous k, same as Bp): M = lane&15, element e -> K = k0 + 16*(lane>>4) + e.
// C/D layout: VGPR r, lane -> row = r + 8*(lane>>4), col = lane&15.
// Requires nrows % 16 == 0 (50000 = 3125*16) and ncols % (16*CT) == 0 (128, 64).

template <int CT>
__global__ void gemm_bf16_wmma(const float* __restrict__ X, const __bf16* __restrict__ Bp,
                               const float* __restrict__ bias, float* __restrict__ Y,
                               int nrows, int ncols) {
  const int K = 128;
  int wid = (int)(((long long)blockIdx.x * blockDim.x + threadIdx.x) >> 5);
  int ngroups = (ncols >> 4) / CT;
  int total = (nrows >> 4) * ngroups;
  if (wid >= total) return;               // whole-wave exit: EXEC stays all-1s for WMMA

  int tm = wid / ngroups;
  int tg = wid - tm * ngroups;
  int lane = threadIdx.x & 31;
  int half = lane >> 4;
  int mr   = lane & 15;

  const float* xrow = X + (size_t)(tm * 16 + mr) * K;

  v8f acc[CT];
#pragma unroll
  for (int j = 0; j < CT; ++j) acc[j] = (v8f){};

#pragma unroll
  for (int ks = 0; ks < KSTEPS; ++ks) {
    v16bf a;
    int lowK = ks * 32 + half * 16;
#pragma unroll
    for (int e = 0; e < 8; ++e) {
      a[e]     = (__bf16)xrow[lowK + e];
      a[e + 8] = (__bf16)xrow[lowK + 8 + e];
    }
#pragma unroll
    for (int j = 0; j < CT; ++j) {
      int tn = tg * CT + j;
      v16bf b = *(const v16bf*)(Bp + ((size_t)(tn * KSTEPS + ks) * 32 + lane) * 16);
      acc[j] = __builtin_amdgcn_wmma_f32_16x16x32_bf16(
          /*neg_a=*/false, a, /*neg_b=*/false, b,
          /*c_mod=*/(short)0, acc[j], /*reuse_a=*/false, /*reuse_b=*/false);
    }
  }

#pragma unroll
  for (int j = 0; j < CT; ++j) {
    int col = (tg * CT + j) * 16 + mr;
    float bv = bias ? bias[col] : 0.f;
#pragma unroll
    for (int r = 0; r < 8; ++r) {
      int row = tm * 16 + r + 8 * half;
      Y[(size_t)row * ncols + col] = acc[j][r] + bv;
    }
  }
}

// ---------------- BatchNorm statistics + apply ----------------

__global__ void colstats_k(const float* __restrict__ Y, float* __restrict__ sums,
                           float* __restrict__ sumsq, int nrows) {
  int c  = threadIdx.x;                 // 128 threads = 128 columns
  int r0 = blockIdx.x * 256;
  int re = min(r0 + 256, nrows);
  float s = 0.f, s2 = 0.f;
  for (int r = r0; r < re; ++r) {
    float v = Y[(size_t)r * 128 + c];
    s += v;
    s2 += v * v;
  }
  atomicAdd(&sums[c], s);
  atomicAdd(&sumsq[c], s2);
}

__global__ void bn_relu_k(const float* __restrict__ Y, const float* __restrict__ sums,
                          const float* __restrict__ sumsq, const float* __restrict__ gamma,
                          const float* __restrict__ beta, float* __restrict__ Xo,
                          long long n4, float invN) {
  long long i = (long long)blockIdx.x * blockDim.x + threadIdx.x;
  if (i >= n4) return;
  float4 v = ((const float4*)Y)[i];
  float o[4] = {v.x, v.y, v.z, v.w};
  int c0 = (int)((i * 4) & 127);
#pragma unroll
  for (int j = 0; j < 4; ++j) {
    int c = c0 + j;
    float mean = sums[c] * invN;
    float var  = sumsq[c] * invN - mean * mean;
    float a    = gamma[c] * rsqrtf(var + 1e-5f);
    o[j] = fmaxf(0.f, (o[j] - mean) * a + beta[c]);
  }
  ((float4*)Xo)[i] = make_float4(o[0], o[1], o[2], o[3]);
}

// ---------------- driver ----------------

extern "C" void kernel_launch(void* const* d_in, const int* in_sizes, int n_in,
                              void* d_out, int out_size, void* d_ws, size_t ws_size,
                              hipStream_t stream) {
  const float* feat = (const float*)d_in[0];
  const int*   src  = (const int*)d_in[1];
  const int*   dst  = (const int*)d_in[2];
  const float* Wl[3] = {(const float*)d_in[3], (const float*)d_in[4], (const float*)d_in[5]};
  const float* gl[3] = {(const float*)d_in[6], (const float*)d_in[8], (const float*)d_in[10]};
  const float* bl[3] = {(const float*)d_in[7], (const float*)d_in[9], (const float*)d_in[11]};
  const float* clsW = (const float*)d_in[12];
  const float* clsb = (const float*)d_in[13];

  const int F = 128, H = 128;
  int N = in_sizes[0] / F;      // 50000
  int E = in_sizes[1];          // 800000
  int C = in_sizes[13];         // 64

  float* ws    = (float*)d_ws;
  float* dinvS = ws;  ws += N;
  float* dinvD = ws;  ws += N;
  float* Xb    = ws;  ws += (size_t)N * H;
  float* Mb    = ws;  ws += (size_t)N * H;
  float* Yb    = ws;  ws += (size_t)N * H;
  float* sums  = ws;  ws += H;             // sums(128) + sumsq(128) contiguous
  float* sumsq = ws;  ws += H;
  __bf16* Bp   = (__bf16*)ws;              // packed B operand, <= 128*128 bf16 = 32 KB
                                           // (byte offset is 32B-aligned: all chunks are)

  // Degrees -> D^{-1/2}
  {
    long long n4 = (long long)(2 * N) / 4;                 // dinvS,dinvD contiguous
    zero_f4<<<(unsigned)((n4 + 255) / 256), 256, 0, stream>>>(dinvS, n4);
    degree_k<<<(E + 255) / 256, 256, 0, stream>>>(src, dst, dinvS, dinvD, E);
    dinv_k<<<(N + 255) / 256, 256, 0, stream>>>(dinvS, N);
    dinv_k<<<(N + 255) / 256, 256, 0, stream>>>(dinvD, N);
  }

  const float* hin = feat;
  long long nh4 = (long long)N * H / 4;
  unsigned zb   = (unsigned)((nh4 + 255) / 256);

  const int CT = 4;                                        // 16x64 strip per wave
  int packH  = (H >> 4) * KSTEPS * 32 * 16;                // 16384 elements
  int gemmWaves = (N / 16) * ((H / 16) / CT);              // 6250
  unsigned gb   = (unsigned)((gemmWaves + 7) / 8);         // 8 waves / block (256 thr)

  for (int l = 0; l < 3; ++l) {
    zero_f4<<<zb, 256, 0, stream>>>(Mb, nh4);
    scatter_k<<<(E + 7) / 8, 256, 0, stream>>>(hin, src, dst, dinvS, dinvD, Mb, E);
    packB_k<<<(packH + 255) / 256, 256, 0, stream>>>(Wl[l], Bp, H);
    gemm_bf16_wmma<CT><<<gb, 256, 0, stream>>>(Mb, Bp, nullptr, Yb, N, H);
    zero_f4<<<1, 64, 0, stream>>>(sums, (2 * H) / 4);
    colstats_k<<<(N + 255) / 256, 128, 0, stream>>>(Yb, sums, sumsq, N);
    bn_relu_k<<<zb, 256, 0, stream>>>(Yb, sums, sumsq, gl[l], bl[l], Xb, nh4, 1.0f / (float)N);
    hin = Xb;
  }

  int packC = (C >> 4) * KSTEPS * 32 * 16;                 // 8192 elements
  packB_k<<<(packC + 255) / 256, 256, 0, stream>>>(clsW, Bp, C);
  int clsWaves = (N / 16) * ((C / 16) / CT);               // 3125
  unsigned cb  = (unsigned)((clsWaves + 7) / 8);
  gemm_bf16_wmma<CT><<<cb, 256, 0, stream>>>(Xb, Bp, clsb, (float*)d_out, N, C);
}
